// Mamba_Model_34677565948029
// MI455X (gfx1250) — matrix operand order repaired and checked by reading, not compiled, not run on hardware
//
#include <hip/hip_runtime.h>
#include <hip/hip_bf16.h>
#include <math.h>

#define B_  4
#define L_  1024
#define DM  512
#define DS  16
#define DFF 2048
#define NL  2
#define DTR 32
#define MR  (B_ * L_)

#define USE_ASYNC 1   // gfx1250 GLOBAL_LOAD_ASYNC_TO_LDS staging

#define LDA 40        // ldsA row stride (u16): 32 + 8 pad, 80B (16B-aligned)
#define LDB 72        // ldsB row stride (u16): 64 + 8 pad, 144B (16B-aligned)

typedef __attribute__((ext_vector_type(16))) __bf16 v16bf;
typedef __attribute__((ext_vector_type(8)))  float  v8f;

__device__ __forceinline__ float siluf(float x) { return x / (1.0f + __expf(-x)); }
__device__ __forceinline__ float softplusf(float x) {
  return fmaxf(x, 0.0f) + log1pf(__expf(-fabsf(x)));
}

// ---------------------------------------------------------------------------
// f32 -> bf16 (round-to-nearest-even) compact copy; optional flip along L and
// column slice (in_stride/col_off) so GEMM A/B operands are always compact.
// ---------------------------------------------------------------------------
__global__ void k_convert_bf16(const float* __restrict__ in, int in_stride, int col_off,
                               unsigned short* __restrict__ out, int rows, int cols,
                               int flip) {
  int i = blockIdx.x * blockDim.x + threadIdx.x;
  if (i >= rows * cols) return;
  int r = i / cols, c = i - r * cols;
  int sr = r;
  if (flip) { int b = r / L_, l = r - b * L_; sr = b * L_ + (L_ - 1 - l); }
  float f = in[(size_t)sr * in_stride + col_off + c];
  unsigned u = __float_as_uint(f);
  out[i] = (unsigned short)((u + 0x7FFFu + ((u >> 16) & 1u)) >> 16);
}

// ---------------------------------------------------------------------------
// Tiled WMMA GEMM: C[M,N] = epilogue(A[M,K] @ W[K,N])
// 128 threads = 4 wave32; 64x64 C tile per block; K-step 32 (bf16 WMMA depth).
// Staging: async global->LDS DMA (ASYNCcnt) when USE_ASYNC, else uint4 copies.
// mode: 0=none 1=+bias 2=relu(+bias) 3=softplus(+bias)
// flip: store row b*L + (L-1-l);  accum: C += result (race-free: unique elems)
// ---------------------------------------------------------------------------
__global__ __launch_bounds__(128) void k_gemm_wmma(
    const unsigned short* __restrict__ A,   // [M,K] bf16
    const unsigned short* __restrict__ W,   // [K,N] bf16
    float* __restrict__ C,                  // [M,N] f32
    int N, int K, const float* __restrict__ bias,
    int mode, int flip, int accum) {
  __shared__ unsigned short ldsA[64 * LDA];  // [m][k]
  __shared__ unsigned short ldsB[32 * LDB];  // [k][n] row-major
  const int tid  = threadIdx.x;
  const int lane = tid & 31;
  const int wave = tid >> 5;
  const int hi   = (lane >> 4) & 1;          // lane half selects K-sub-run
  const int l15  = lane & 15;
  const int bM   = blockIdx.y * 64;
  const int bN   = blockIdx.x * 64;

  // this thread's two 16B staging chunks (straight-line, EXEC stays full)
  const int ca0 = tid, ca1 = tid + 128;            // A: 256 chunks of 8 u16
  const int am0 = ca0 >> 2, ak0 = (ca0 & 3) << 3;
  const int am1 = ca1 >> 2, ak1 = (ca1 & 3) << 3;
  const int bk0 = ca0 >> 3, bn0 = (ca0 & 7) << 3;  // B: 256 chunks of 8 u16
  const int bk1 = ca1 >> 3, bn1 = (ca1 & 7) << 3;

  v8f acc[4] = {};

  for (int k0 = 0; k0 < K; k0 += 32) {
#if USE_ASYNC
    {
      unsigned la0 = (unsigned)(size_t)&ldsA[am0 * LDA + ak0];
      unsigned la1 = (unsigned)(size_t)&ldsA[am1 * LDA + ak1];
      unsigned lb0 = (unsigned)(size_t)&ldsB[bk0 * LDB + bn0];
      unsigned lb1 = (unsigned)(size_t)&ldsB[bk1 * LDB + bn1];
      const unsigned short* ga0 = A + (size_t)(bM + am0) * K + k0 + ak0;
      const unsigned short* ga1 = A + (size_t)(bM + am1) * K + k0 + ak1;
      const unsigned short* gb0 = W + (size_t)(k0 + bk0) * N + bN + bn0;
      const unsigned short* gb1 = W + (size_t)(k0 + bk1) * N + bN + bn1;
      asm volatile("global_load_async_to_lds_b128 %0, %1, off"
                   :: "v"(la0), "v"(ga0) : "memory");
      asm volatile("global_load_async_to_lds_b128 %0, %1, off"
                   :: "v"(la1), "v"(ga1) : "memory");
      asm volatile("global_load_async_to_lds_b128 %0, %1, off"
                   :: "v"(lb0), "v"(gb0) : "memory");
      asm volatile("global_load_async_to_lds_b128 %0, %1, off"
                   :: "v"(lb1), "v"(gb1) : "memory");
      asm volatile("s_wait_asynccnt 0x0" ::: "memory");
    }
#else
    *(uint4*)(&ldsA[am0 * LDA + ak0]) =
        *(const uint4*)(A + (size_t)(bM + am0) * K + k0 + ak0);
    *(uint4*)(&ldsA[am1 * LDA + ak1]) =
        *(const uint4*)(A + (size_t)(bM + am1) * K + k0 + ak1);
    *(uint4*)(&ldsB[bk0 * LDB + bn0]) =
        *(const uint4*)(W + (size_t)(k0 + bk0) * N + bN + bn0);
    *(uint4*)(&ldsB[bk1 * LDB + bn1]) =
        *(const uint4*)(W + (size_t)(k0 + bk1) * N + bN + bn1);
#endif
    if (k0 + 32 < K) {
      __builtin_prefetch(A + (size_t)(bM + (tid >> 1)) * K + k0 + 32, 0, 0);
      __builtin_prefetch(W + (size_t)(k0 + 32 + (tid & 31)) * N + bN, 0, 0);
    }
    __syncthreads();

    // B fragment: lane = column n (wave*16 + l15); contiguous 16 K values
    // (lanes 0-15: K 0..15, lanes 16-31: K 16..31), gathered from [k][n] LDS.
    union { v16bf v; unsigned u[8]; } bf;
    {
      const unsigned short* pb = ldsB + (hi ? 16 * LDB : 0) + wave * 16 + l15;
      #pragma unroll
      for (int j = 0; j < 8; ++j) {
        unsigned lo = pb[(2 * j) * LDB];
        unsigned hh = pb[(2 * j + 1) * LDB];
        bf.u[j] = lo | (hh << 16);
      }
    }
    // A fragments: lane = row m; K runs 0..15 (lanes 0-15) /
    // 16..31 (lanes 16-31); pairs contiguous -> u32 loads.
    #pragma unroll
    for (int mt = 0; mt < 4; ++mt) {
      union { v16bf v; unsigned u[8]; } af;
      const unsigned* p = (const unsigned*)(ldsA + (mt * 16 + l15) * LDA);
      int kb = hi ? 8 : 0;  // u32 units
      #pragma unroll
      for (int j = 0; j < 4; ++j) af.u[j] = p[kb + j];
      #pragma unroll
      for (int j = 0; j < 4; ++j) af.u[4 + j] = p[kb + 4 + j];
      acc[mt] = __builtin_amdgcn_wmma_f32_16x16x32_bf16(
          false, af.v, false, bf.v, (short)0, acc[mt], false, false);
    }
    __syncthreads();
  }

  const int col = bN + wave * 16 + l15;
  const float bv = (bias != nullptr) ? bias[col] : 0.0f;
  #pragma unroll
  for (int mt = 0; mt < 4; ++mt) {
    #pragma unroll
    for (int r = 0; r < 8; ++r) {
      int row = bM + mt * 16 + r + (hi ? 8 : 0);
      float v = acc[mt][r];
      if (mode >= 1) v += bv;
      if (mode == 2) v = fmaxf(v, 0.0f);
      if (mode == 3) v = softplusf(v);
      int orow = row;
      if (flip) { int b = row / L_, l = row - b * L_; orow = b * L_ + (L_ - 1 - l); }
      size_t off = (size_t)orow * N + col;
      if (accum) v += C[off];
      C[off] = v;
    }
  }
}

// causal depthwise conv (k=2) + SiLU; xi = xz[:, :DM] (row stride 2*DM)
__global__ void k_conv_silu(const float* __restrict__ xz, const float* __restrict__ cw,
                            const float* __restrict__ cb, float* __restrict__ xc) {
  int i = blockIdx.x * blockDim.x + threadIdx.x;
  if (i >= MR * DM) return;
  int r = i / DM, d = i - r * DM;
  int l = r & (L_ - 1);
  float xcur  = xz[(size_t)r * (2 * DM) + d];
  float xprev = (l > 0) ? xz[(size_t)(r - 1) * (2 * DM) + d] : 0.0f;
  xc[i] = siluf(xprev * cw[d * 2 + 0] + xcur * cw[d * 2 + 1] + cb[d]);
}

// selective scan: one thread per (b,d), n=16 state in registers
__global__ void k_scan(const float* __restrict__ delta, const float* __restrict__ u,
                       const float* __restrict__ xdbl, const float* __restrict__ A_log,
                       const float* __restrict__ Dp, float* __restrict__ y) {
  int b = blockIdx.x;
  int d = threadIdx.x;  // blockDim = 512
  float Av[DS], h[DS];
  #pragma unroll
  for (int n = 0; n < DS; ++n) { Av[n] = -__expf(A_log[d * DS + n]); h[n] = 0.0f; }
  float Dd = Dp[d];
  for (int t = 0; t < L_; ++t) {
    size_t r = (size_t)b * L_ + t;
    float dl = delta[r * DM + d];
    float uu = u[r * DM + d];
    const float* Bp = xdbl + r * 64 + DTR;  // cols 32..47
    const float* Cp = Bp + DS;              // cols 48..63
    float acc = uu * Dd;
    float dbu = dl * uu;
    #pragma unroll
    for (int n = 0; n < DS; ++n) {
      h[n] = __expf(dl * Av[n]) * h[n] + dbu * Bp[n];
      acc += h[n] * Cp[n];
    }
    y[r * DM + d] = acc;
  }
}

// y *= silu(z), z = xz[:, DM:2*DM]
__global__ void k_gate(float* __restrict__ y, const float* __restrict__ xz) {
  int i = blockIdx.x * blockDim.x + threadIdx.x;
  if (i >= MR * DM) return;
  int r = i / DM, d = i - r * DM;
  y[i] *= siluf(xz[(size_t)r * (2 * DM) + DM + d]);
}

// out = LayerNorm(a (+ b)) * w + bias ; block per row, 256 threads x 2 elems
__global__ void k_add_ln(const float* __restrict__ a, const float* __restrict__ b,
                         const float* __restrict__ w, const float* __restrict__ bias,
                         float* __restrict__ out) {
  __shared__ float red[256];
  int row = blockIdx.x, t = threadIdx.x;
  size_t off = (size_t)row * DM;
  float v0 = a[off + t], v1 = a[off + t + 256];
  if (b != nullptr) { v0 += b[off + t]; v1 += b[off + t + 256]; }
  red[t] = v0 + v1;
  __syncthreads();
  for (int o = 128; o > 0; o >>= 1) { if (t < o) red[t] += red[t + o]; __syncthreads(); }
  float mu = red[0] * (1.0f / DM);
  __syncthreads();
  float d0 = v0 - mu, d1 = v1 - mu;
  red[t] = d0 * d0 + d1 * d1;
  __syncthreads();
  for (int o = 128; o > 0; o >>= 1) { if (t < o) red[t] += red[t + o]; __syncthreads(); }
  float rstd = rsqrtf(red[0] * (1.0f / DM) + 1e-5f);
  out[off + t]       = d0 * rstd * w[t] + bias[t];
  out[off + t + 256] = d1 * rstd * w[t + 256] + bias[t + 256];
}

// ---------------------------------------------------------------------------
extern "C" void kernel_launch(void* const* d_in, const int* in_sizes, int n_in,
                              void* d_out, int out_size, void* d_ws, size_t ws_size,
                              hipStream_t stream) {
  (void)in_sizes; (void)n_in; (void)out_size; (void)ws_size;
  const float* x_enc = (const float*)d_in[0];
  int idx = 1;

  char* base = (char*)d_ws;
  auto carve = [&](size_t bytes) {
    char* p = base; base += (bytes + 255) & ~(size_t)255; return (void*)p;
  };
  float* cur_x = (float*)carve((size_t)MR * DM * 4);
  float* new_x = (float*)carve((size_t)MR * DM * 4);
  float* xz    = (float*)carve((size_t)MR * 2 * DM * 4);
  float* xc    = (float*)carve((size_t)MR * DM * 4);
  float* xdbl  = (float*)carve((size_t)MR * 64 * 4);
  float* delta = (float*)carve((size_t)MR * DM * 4);  // reused as FFN output
  float* yscan = (float*)carve((size_t)MR * DM * 4);
  float* x1    = (float*)carve((size_t)MR * DM * 4);
  float* ffh   = (float*)carve((size_t)MR * DFF * 4);
  unsigned short* a_bf = (unsigned short*)carve((size_t)MR * DFF * 2);
  unsigned short* w_bf = (unsigned short*)carve((size_t)DFF * DM * 2);

  auto cvt = [&](const float* in, int stride, int coff, unsigned short* out,
                 int rows, int cols, int flip) {
    int tot = rows * cols;
    k_convert_bf16<<<(tot + 255) / 256, 256, 0, stream>>>(in, stride, coff, out,
                                                          rows, cols, flip);
  };
  auto gemm = [&](const unsigned short* Ab, const unsigned short* Wb, float* Cc,
                  int K, int N, const float* bias, int mode, int flip, int accum) {
    dim3 grid(N / 64, MR / 64);
    k_gemm_wmma<<<grid, 128, 0, stream>>>(Ab, Wb, Cc, N, K, bias, mode, flip, accum);
  };

  hipMemcpyAsync(cur_x, x_enc, (size_t)MR * DM * 4, hipMemcpyDeviceToDevice, stream);

  for (int li = 0; li < NL; ++li) {
    const float* mp[2][9];
    for (int dir = 0; dir < 2; ++dir)
      for (int j = 0; j < 9; ++j) mp[dir][j] = (const float*)d_in[idx++];
    const float* ln1w = (const float*)d_in[idx++];
    const float* ln1b = (const float*)d_in[idx++];
    const float* ln2w = (const float*)d_in[idx++];
    const float* ln2b = (const float*)d_in[idx++];
    const float* ff1w = (const float*)d_in[idx++];
    const float* ff1b = (const float*)d_in[idx++];
    const float* ff2w = (const float*)d_in[idx++];
    const float* ff2b = (const float*)d_in[idx++];

    for (int dir = 0; dir < 2; ++dir) {
      const float* in_proj  = mp[dir][0];
      const float* conv_w   = mp[dir][1];
      const float* conv_b   = mp[dir][2];
      const float* x_proj   = mp[dir][3];
      const float* dt_w     = mp[dir][4];
      const float* dt_b     = mp[dir][5];
      const float* A_log    = mp[dir][6];
      const float* Dparam   = mp[dir][7];
      const float* out_proj = mp[dir][8];
      const int rev = dir;  // dir 1 = reversed sequence

      // xz = (flip?) x @ in_proj   [4096, 1024]
      cvt(cur_x, DM, 0, a_bf, MR, DM, rev);
      cvt(in_proj, 2 * DM, 0, w_bf, DM, 2 * DM, 0);
      gemm(a_bf, w_bf, xz, DM, 2 * DM, nullptr, 0, 0, 0);

      // xc = silu(causal depthwise conv(xi))
      { int tot = MR * DM;
        k_conv_silu<<<(tot + 255) / 256, 256, 0, stream>>>(xz, conv_w, conv_b, xc); }

      // x_dbl = xc @ x_proj   [4096, 64]
      cvt(xc, DM, 0, a_bf, MR, DM, 0);
      cvt(x_proj, 64, 0, w_bf, DM, 64, 0);
      gemm(a_bf, w_bf, xdbl, DM, 64, nullptr, 0, 0, 0);

      // delta = softplus(dt @ dt_w + dt_b)   [4096, 512]
      cvt(xdbl, 64, 0, a_bf, MR, DTR, 0);
      cvt(dt_w, DM, 0, w_bf, DTR, DM, 0);
      gemm(a_bf, w_bf, delta, DTR, DM, dt_b, 3, 0, 0);

      // selective scan + gate
      k_scan<<<B_, DM, 0, stream>>>(delta, xc, xdbl, A_log, Dparam, yscan);
      { int tot = MR * DM;
        k_gate<<<(tot + 255) / 256, 256, 0, stream>>>(yscan, xz); }

      // new_x (+)= unflip( gated @ out_proj )
      cvt(yscan, DM, 0, a_bf, MR, DM, 0);
      cvt(out_proj, DM, 0, w_bf, DM, DM, 0);
      gemm(a_bf, w_bf, new_x, DM, DM, nullptr, 0, rev, rev);
    }

    // x1 = LN(x + new_x)
    k_add_ln<<<MR, 256, 0, stream>>>(cur_x, new_x, ln1w, ln1b, x1);

    // FFN
    cvt(x1, DM, 0, a_bf, MR, DM, 0);
    cvt(ff1w, DFF, 0, w_bf, DM, DFF, 0);
    gemm(a_bf, w_bf, ffh, DM, DFF, ff1b, 2, 0, 0);
    cvt(ffh, DFF, 0, a_bf, MR, DFF, 0);
    cvt(ff2w, DM, 0, w_bf, DFF, DM, 0);
    float* ffo = delta;
    gemm(a_bf, w_bf, ffo, DFF, DM, ff2b, 1, 0, 0);

    // x = LN(x1 + ffo)
    k_add_ln<<<MR, 256, 0, stream>>>(x1, ffo, ln2w, ln2b, cur_x);
  }

  const float* nw = (const float*)d_in[idx++];
  const float* nb = (const float*)d_in[idx++];
  k_add_ln<<<MR, 256, 0, stream>>>(cur_x, nullptr, nw, nb, (float*)d_out);
}
